// SelfAttentionDatasetEncoder_79328045957410
// MI455X (gfx1250) — 16-bit and 8-bit matrix operand order repaired and checked by reading, compiled, not run on hardware
//
#include <hip/hip_runtime.h>
#include <hip/hip_bf16.h>

typedef __bf16 bf16_t;
typedef __attribute__((ext_vector_type(16))) __bf16 v16bf;
typedef __attribute__((ext_vector_type(8)))  __bf16 v8bf;
typedef __attribute__((ext_vector_type(8)))  float  v8f;
typedef __attribute__((ext_vector_type(8)))  int    v8i;
typedef unsigned int  u32;
typedef unsigned char u8;
typedef unsigned long long u64;

#define G_    64
#define N_    1024
#define DIN_  128
#define DOUT_ 256
#define JT_   (N_ / 16)
#define QPAD_ 272

__device__ inline u32 f32_to_e4m3(float x)
{
    u32 u = __float_as_uint(x);
    u32 sign = (u >> 24) & 0x80u;
    u32 au = u & 0x7FFFFFFFu;
    if (au >= 0x43E00000u) return sign | 0x7Eu;
    if (au <  0x3C800000u) return sign;
    u32 lsb = (au >> 20) & 1u;
    au += 0x0007FFFFu + lsb;
    u32 exp = (au >> 23) - 127u + 7u;
    u32 man = (au >> 20) & 7u;
    return sign | (exp << 3) | man;
}

__global__ __launch_bounds__(256)
void prep_w_kernel(const float* __restrict__ Wq, const float* __restrict__ Wk,
                   bf16_t* __restrict__ Wqt, bf16_t* __restrict__ Wkt)
{
    const int idx = blockIdx.x * 256 + threadIdx.x;
    const int c = idx >> 7;
    const int r = idx & (DIN_ - 1);
    Wqt[idx] = (bf16_t)Wq[(size_t)r * DOUT_ + c];
    Wkt[idx] = (bf16_t)Wk[(size_t)r * DOUT_ + c];
}

__global__ __launch_bounds__(256)
void proj_qk_kernel(const float* __restrict__ X,
                    const bf16_t* __restrict__ Wqt, const float* __restrict__ bq,
                    const bf16_t* __restrict__ Wkt, const float* __restrict__ bk,
                    u8* __restrict__ Qf8, u8* __restrict__ Kf8)
{
    __shared__ bf16_t Xlds[16][DIN_ + 8];
    __shared__ float  Dlds[2][16][DOUT_ + 4];

    const int g   = blockIdx.x;
    const int i0  = blockIdx.y * 16;
    const int tid = threadIdx.x;

    const float* Xg = X + ((size_t)g * N_ + i0) * DIN_;
    for (int e = tid; e < 16 * DIN_; e += 256)
        Xlds[e >> 7][e & 127] = (bf16_t)Xg[e];
    __syncthreads();

    const int wave = tid >> 5;
    const int lane = tid & 31;
    const int m    = lane & 15;
    const int h    = lane >> 4;

    v16bf a[4];
    #pragma unroll
    for (int kb = 0; kb < 4; ++kb) {
        v8bf lo = *(const v8bf*)&Xlds[m][kb * 32 + h * 8];
        v8bf hi = *(const v8bf*)&Xlds[m][kb * 32 + 16 + h * 8];
        #pragma unroll
        for (int e = 0; e < 8; ++e) { a[kb][e] = lo[e]; a[kb][e + 8] = hi[e]; }
    }

    #pragma unroll
    for (int sel = 0; sel < 4; ++sel) {
        const int mat = sel >> 1;
        const int n0  = (wave + (sel & 1) * 8) * 16;
        const bf16_t* Wt = mat ? Wkt : Wqt;
        const float*  bb = mat ? bk  : bq;

        v8f c = {};
        #pragma unroll
        for (int kb = 0; kb < 4; ++kb) {
            const bf16_t* wp = Wt + (size_t)(n0 + m) * DIN_ + kb * 32 + h * 8;
            v8bf blo = *(const v8bf*)(wp);
            v8bf bhi = *(const v8bf*)(wp + 16);
            v16bf bm;
            #pragma unroll
            for (int e = 0; e < 8; ++e) { bm[e] = blo[e]; bm[e + 8] = bhi[e]; }
            c = __builtin_amdgcn_wmma_f32_16x16x32_bf16(
                    false, a[kb], false, bm, (short)0, c, false, false);
        }
        const float bias = bb[n0 + m];
        #pragma unroll
        for (int r = 0; r < 8; ++r)
            Dlds[mat][r + h * 8][n0 + m] = c[r] + bias;
    }
    __syncthreads();

    #pragma unroll
    for (int mat = 0; mat < 2; ++mat) {
        u8* outp = (mat ? Kf8 : Qf8) + ((size_t)g * N_ + i0) * DOUT_;
        for (int ch = tid; ch < 512; ch += 256) {
            const int row = ch >> 5;
            const int c8  = (ch & 31) * 8;
            const float* src = &Dlds[mat][row][c8];
            uint2 v;
            v.x = f32_to_e4m3(src[0])        | (f32_to_e4m3(src[1]) << 8)
                | (f32_to_e4m3(src[2]) << 16) | (f32_to_e4m3(src[3]) << 24);
            v.y = f32_to_e4m3(src[4])        | (f32_to_e4m3(src[5]) << 8)
                | (f32_to_e4m3(src[6]) << 16) | (f32_to_e4m3(src[7]) << 24);
            *(uint2*)(outp + (size_t)row * DOUT_ + c8) = v;
        }
    }
}

__global__ __launch_bounds__(256)
void attn_row_kernel(const u8* __restrict__ Qf8,
                     const u8* __restrict__ Kf8,
                     float* __restrict__ rowOut)
{
    __shared__ u8    Qlds[32][QPAD_];
    __shared__ float Slds[32][N_ + 4];

    const int g    = blockIdx.x;
    const int i0   = blockIdx.y * 32;
    const int tid  = threadIdx.x;
    const int wave = tid >> 5;
    const int lane = tid & 31;
    const int m    = lane & 15;
    const int h    = lane >> 4;

    {
        const u8* Qg = Qf8 + ((size_t)g * N_ + i0) * DOUT_;
        for (int ch = tid; ch < 512; ch += 256) {
            const int row = ch >> 4;
            const int cb  = (ch & 15) * 16;
            *(uint4*)&Qlds[row][cb] = *(const uint4*)(Qg + (size_t)row * DOUT_ + cb);
        }
    }
    __syncthreads();

    const u8* Kg = Kf8 + (size_t)g * N_ * DOUT_;

    for (int jt = wave; jt < JT_; jt += 8) {
        if (jt + 8 < JT_) {
            const char* p = (const char*)(Kg + (size_t)((jt + 8) * 16 + m) * DOUT_);
            __builtin_prefetch(p, 0, 3);
            __builtin_prefetch(p + 128, 0, 3);
        }
        const u8* Kr = Kg + (size_t)(jt * 16 + m) * DOUT_;
        v8f c0 = {}, c1 = {};
        #pragma unroll
        for (int s = 0; s < 4; ++s) {
            uint4 b0 = *(const uint4*)(Kr + s * 64 + h * 16);
            uint4 b1 = *(const uint4*)(Kr + s * 64 + 32 + h * 16);
            v8i bf;
            bf[0] = (int)b0.x; bf[1] = (int)b0.y; bf[2] = (int)b0.z; bf[3] = (int)b0.w;
            bf[4] = (int)b1.x; bf[5] = (int)b1.y; bf[6] = (int)b1.z; bf[7] = (int)b1.w;

            v8i a0, a1;
            #pragma unroll
            for (int it = 0; it < 2; ++it) {
                const u8* qr = &Qlds[it * 16 + m][s * 64 + h * 16];
                u64 q0 = *(const u64*)(qr + 0);
                u64 q1 = *(const u64*)(qr + 8);
                u64 q2 = *(const u64*)(qr + 32);
                u64 q3 = *(const u64*)(qr + 40);
                v8i af;
                af[0] = (int)(u32)q0; af[1] = (int)(u32)(q0 >> 32);
                af[2] = (int)(u32)q1; af[3] = (int)(u32)(q1 >> 32);
                af[4] = (int)(u32)q2; af[5] = (int)(u32)(q2 >> 32);
                af[6] = (int)(u32)q3; af[7] = (int)(u32)(q3 >> 32);
                if (it == 0) a0 = af; else a1 = af;
            }
            c0 = __builtin_amdgcn_wmma_f32_16x16x64_fp8_fp8(
                    a0, bf, (short)0, c0, false, false);
            c1 = __builtin_amdgcn_wmma_f32_16x16x64_fp8_fp8(
                    a1, bf, (short)0, c1, false, false);
        }
        #pragma unroll
        for (int r = 0; r < 8; ++r) {
            Slds[r + h * 8][jt * 16 + m]      = c0[r];
            Slds[16 + r + h * 8][jt * 16 + m] = c1[r];
        }
    }
    __syncthreads();

    #pragma unroll
    for (int rr = 0; rr < 4; ++rr) {
        const int row = wave * 4 + rr;
        float mx = -3.0e38f;
        for (int j = lane; j < N_; j += 32) mx = fmaxf(mx, Slds[row][j]);
        #pragma unroll
        for (int mask = 1; mask < 32; mask <<= 1)
            mx = fmaxf(mx, __shfl_xor(mx, mask, 32));
        float sm = 0.0f;
        for (int j = lane; j < N_; j += 32) sm += __expf(Slds[row][j] - mx);
        #pragma unroll
        for (int mask = 1; mask < 32; mask <<= 1)
            sm += __shfl_xor(sm, mask, 32);
        float t = 0.0f;
        for (int j = lane; j < N_; j += 32) t += __expf(Slds[row][j] - mx) / sm;
        #pragma unroll
        for (int mask = 1; mask < 32; mask <<= 1)
            t += __shfl_xor(t, mask, 32);
        if (lane == 0)
            rowOut[(size_t)g * N_ + i0 + row] = t;
    }
}

__global__ __launch_bounds__(256)
void out_kernel(const float* __restrict__ X,
                const float* __restrict__ Wv, const float* __restrict__ bv,
                const float* __restrict__ row, float* __restrict__ out)
{
    __shared__ float y2[2][DIN_];
    __shared__ float rs2[2];
    __shared__ float yl[DIN_];
    __shared__ float rl;

    const int g    = blockIdx.x;
    const int tid  = threadIdx.x;
    const int d    = tid & (DIN_ - 1);
    const int half = tid >> 7;

    const float* Xg = X + (size_t)g * N_ * DIN_;
    const float* rg = row + (size_t)g * N_;

    float y = 0.0f, rs = 0.0f;
    for (int i = half; i < N_; i += 2) {
        const float w = rg[i];
        y += w * Xg[(size_t)i * DIN_ + d];
        rs += w;
    }
    y2[half][d] = y;
    if (d == 0) rs2[half] = rs;
    __syncthreads();
    if (tid < DIN_) yl[tid] = y2[0][tid] + y2[1][tid];
    if (tid == 0)   rl = rs2[0] + rs2[1];
    __syncthreads();

    float acc = 0.0f;
    for (int c = 0; c < DIN_; ++c)
        acc += yl[c] * Wv[(size_t)c * DOUT_ + tid];
    out[(size_t)g * DOUT_ + tid] = (acc + rl * bv[tid]) * (1.0f / (float)N_);
}

extern "C" void kernel_launch(void* const* d_in, const int* in_sizes, int n_in,
                              void* d_out, int out_size, void* d_ws, size_t ws_size,
                              hipStream_t stream)
{
    (void)in_sizes; (void)n_in; (void)out_size; (void)ws_size;
    const float* X  = (const float*)d_in[0];
    const float* Wq = (const float*)d_in[1];
    const float* bq = (const float*)d_in[2];
    const float* Wk = (const float*)d_in[3];
    const float* bk = (const float*)d_in[4];
    const float* Wv = (const float*)d_in[5];
    const float* bv = (const float*)d_in[6];
    float* out = (float*)d_out;

    char* ws = (char*)d_ws;
    const size_t qk_bytes = (size_t)G_ * N_ * DOUT_;
    u8*    Qf8 = (u8*)ws;
    u8*    Kf8 = (u8*)(ws + qk_bytes);
    float* row = (float*)(ws + 2 * qk_bytes);
    bf16_t* Wqt = (bf16_t*)(ws + 2 * qk_bytes + (size_t)G_ * N_ * sizeof(float));
    bf16_t* Wkt = Wqt + (size_t)DIN_ * DOUT_;

    prep_w_kernel<<<(DIN_ * DOUT_) / 256, 256, 0, stream>>>(Wq, Wk, Wqt, Wkt);
    dim3 gridP(G_, N_ / 16);
    proj_qk_kernel<<<gridP, 256, 0, stream>>>(X, Wqt, bq, Wkt, bk, Qf8, Kf8);
    dim3 gridA(G_, N_ / 32);
    attn_row_kernel<<<gridA, 256, 0, stream>>>(Qf8, Kf8, row);
    out_kernel<<<G_, 256, 0, stream>>>(X, Wv, bv, row, out);
}
